// DendriticLayer_20083267076205
// MI455X (gfx1250) — matrix operand order repaired and checked by reading, not compiled, not run on hardware
//
#include <hip/hip_runtime.h>

typedef __attribute__((ext_vector_type(16))) _Float16     v16h;
typedef __attribute__((ext_vector_type(8)))  float        v8f;
typedef __attribute__((ext_vector_type(4)))  _Float16     v4h;
typedef __attribute__((ext_vector_type(4)))  unsigned int v4u;

#define IN_DIM     4096
#define N_SOMA     16384
#define N_NEURONS  1024
#define BATCH      256
#define NEG_SLOPE  0.1f

#define TM   64    // M rows per workgroup
#define TN   128   // N cols per workgroup
#define TK   32    // K per step (WMMA f16 K)
#define LDSK 40    // padded LDS row length in f16 (80B: 16B-aligned, conflict-free b128 reads)

// ---------------------------------------------------------------------------
// Stage 1: Y[256,16384] = lrelu(X @ W1^T + b1), f16 WMMA with f32 accumulate.
// ---------------------------------------------------------------------------
__global__ __launch_bounds__(256) void dendrite_gemm(
    const float* __restrict__ x,      // [BATCH, IN_DIM]
    const float* __restrict__ dW,     // [N_SOMA, IN_DIM] (mask pre-applied)
    const float* __restrict__ dB,     // [N_SOMA]
    float* __restrict__ dend)         // [BATCH, N_SOMA]
{
  __shared__ _Float16 As[2][TM][LDSK];
  __shared__ _Float16 Bs[2][TN][LDSK];

  const int tid   = threadIdx.x;
  const int lane  = tid & 31;
  const int wave  = tid >> 5;       // 0..7
  const int hi    = lane >> 4;      // lane half
  const int mn    = lane & 15;
  const int waveM = wave & 1;       // 2 waves along M
  const int waveN = wave >> 1;      // 4 waves along N
  const int mBase = blockIdx.x * TM;
  const int nBase = blockIdx.y * TN;

  // cooperative tile loaders: 8 threads per 32-wide K row, 4 floats each
  const int lr = tid >> 3;          // 0..31
  const int lc = (tid & 7) * 4;     // 0,4,...,28

  const float* xA = x  + (size_t)(mBase + lr) * IN_DIM + lc;
  const float* wB = dW + (size_t)(nBase + lr) * IN_DIM + lc;

  v8f acc00 = {}; v8f acc01 = {}; v8f acc10 = {}; v8f acc11 = {};

  float4 ar[2];
  float4 br[4];

  auto gload = [&](int k0) {
    ar[0] = *(const float4*)(xA + k0);
    ar[1] = *(const float4*)(xA + (size_t)32 * IN_DIM + k0);
    br[0] = *(const float4*)(wB + k0);
    br[1] = *(const float4*)(wB + (size_t)32 * IN_DIM + k0);
    br[2] = *(const float4*)(wB + (size_t)64 * IN_DIM + k0);
    br[3] = *(const float4*)(wB + (size_t)96 * IN_DIM + k0);
  };

  auto cvt4 = [](float4 f) {
    v4h h;
    h[0] = (_Float16)f.x; h[1] = (_Float16)f.y;
    h[2] = (_Float16)f.z; h[3] = (_Float16)f.w;
    return h;
  };

  auto lstore = [&](int buf) {
    *(v4h*)&As[buf][lr     ][lc] = cvt4(ar[0]);
    *(v4h*)&As[buf][lr + 32][lc] = cvt4(ar[1]);
    *(v4h*)&Bs[buf][lr     ][lc] = cvt4(br[0]);
    *(v4h*)&Bs[buf][lr + 32][lc] = cvt4(br[1]);
    *(v4h*)&Bs[buf][lr + 64][lc] = cvt4(br[2]);
    *(v4h*)&Bs[buf][lr + 96][lc] = cvt4(br[3]);
  };

  auto compute = [&](int buf) {
    const _Float16* a0 = &As[buf][waveM * 32      + mn][0];
    const _Float16* a1 = &As[buf][waveM * 32 + 16 + mn][0];
    const _Float16* b0 = &Bs[buf][waveN * 32      + mn][0];
    const _Float16* b1 = &Bs[buf][waveN * 32 + 16 + mn][0];

    // A fragment (16x32 f16): lane half hi reads K = 8hi..8hi+7 and 16+8hi..23+8hi
    v16h A0, A1, B0, B1;
    ((v4u*)&A0)[0] = *(const v4u*)(a0 + 8 * hi);
    ((v4u*)&A0)[1] = *(const v4u*)(a0 + 16 + 8 * hi);
    ((v4u*)&A1)[0] = *(const v4u*)(a1 + 8 * hi);
    ((v4u*)&A1)[1] = *(const v4u*)(a1 + 16 + 8 * hi);
    // B fragment (32x16 f16): lane half hi reads K = 8hi..8hi+7 and 16+8hi..23+8hi of column mn
    ((v4u*)&B0)[0] = *(const v4u*)(b0 + 8 * hi);
    ((v4u*)&B0)[1] = *(const v4u*)(b0 + 16 + 8 * hi);
    ((v4u*)&B1)[0] = *(const v4u*)(b1 + 8 * hi);
    ((v4u*)&B1)[1] = *(const v4u*)(b1 + 16 + 8 * hi);

    acc00 = __builtin_amdgcn_wmma_f32_16x16x32_f16(false, A0, false, B0, (short)0, acc00, false, false);
    acc01 = __builtin_amdgcn_wmma_f32_16x16x32_f16(false, A0, false, B1, (short)0, acc01, false, false);
    acc10 = __builtin_amdgcn_wmma_f32_16x16x32_f16(false, A1, false, B0, (short)0, acc10, false, false);
    acc11 = __builtin_amdgcn_wmma_f32_16x16x32_f16(false, A1, false, B1, (short)0, acc11, false, false);
  };

  gload(0);
  lstore(0);
  __syncthreads();

  const int NT = IN_DIM / TK;  // 128
#pragma unroll 1
  for (int t = 0; t < NT; ++t) {
    const int nk = (t + 1) * TK;
    if (t + 2 < NT) {  // stream-ahead hint on the 268MB weight stream
      __builtin_prefetch(wB + (size_t)(t + 2) * TK, 0, 1);
      __builtin_prefetch(wB + (size_t)64 * IN_DIM + (size_t)(t + 2) * TK, 0, 1);
    }
    if (nk < IN_DIM) gload(nk);
    compute(t & 1);
    if (nk < IN_DIM) lstore((t + 1) & 1);
    __syncthreads();
  }

  // Epilogue: bias + LeakyReLU, write f32 Y. C layout: VGPR r -> M=r+8hi, N=lane&15.
  const float* accs[4] = {(const float*)&acc00, (const float*)&acc01,
                          (const float*)&acc10, (const float*)&acc11};
#pragma unroll
  for (int im = 0; im < 2; ++im) {
#pragma unroll
    for (int in = 0; in < 2; ++in) {
      const float* a   = accs[im * 2 + in];
      const int nCol   = nBase + waveN * 32 + in * 16 + mn;
      const float bias = dB[nCol];
#pragma unroll
      for (int r = 0; r < 8; ++r) {
        const int mRow = mBase + waveM * 32 + im * 16 + r + 8 * hi;
        float v = a[r] + bias;
        v = (v >= 0.0f) ? v : NEG_SLOPE * v;
        dend[(size_t)mRow * N_SOMA + nCol] = v;
      }
    }
  }
}

// ---------------------------------------------------------------------------
// Stage 2: block-diagonal soma. out[b,n] = lrelu(dot16(Y[b,16n:], W2[n,16n:]) + b2[n])
// Reads only the 64B diagonal block per neuron instead of the dense 64MB W2.
// ---------------------------------------------------------------------------
__global__ __launch_bounds__(256) void soma_kernel(
    const float* __restrict__ dend,   // [BATCH, N_SOMA]
    const float* __restrict__ sW,     // [N_NEURONS, N_SOMA] block-diagonal
    const float* __restrict__ sB,     // [N_NEURONS]
    float* __restrict__ out)          // [BATCH, N_NEURONS]
{
  const int gid = blockIdx.x * 256 + threadIdx.x;  // b*1024 + n
  const int b = gid >> 10;
  const int n = gid & 1023;

  const float4* dp = (const float4*)(dend + (size_t)b * N_SOMA + n * 16);
  const float4* wp = (const float4*)(sW + (size_t)n * N_SOMA + n * 16);

  float s = 0.0f;
#pragma unroll
  for (int j = 0; j < 4; ++j) {
    const float4 d = dp[j];
    const float4 w = wp[j];
    s += d.x * w.x + d.y * w.y + d.z * w.z + d.w * w.w;
  }
  s += sB[n];
  out[gid] = (s >= 0.0f) ? s : NEG_SLOPE * s;
}

// ---------------------------------------------------------------------------
extern "C" void kernel_launch(void* const* d_in, const int* in_sizes, int n_in,
                              void* d_out, int out_size, void* d_ws, size_t ws_size,
                              hipStream_t stream) {
  const float* x  = (const float*)d_in[0];  // [256, 4096]
  const float* dW = (const float*)d_in[1];  // [16384, 4096] (already masked)
  const float* dB = (const float*)d_in[2];  // [16384]
  const float* sW = (const float*)d_in[3];  // [1024, 16384] (already masked)
  const float* sB = (const float*)d_in[4];  // [1024]
  // d_in[5] dendrite_mask, d_in[6] soma_mask: unused (masking is idempotent)

  float* dend = (float*)d_ws;               // [256, 16384] f32 = 16 MB scratch

  dim3 grid1(BATCH / TM, N_SOMA / TN);      // (4, 128); M fastest so W1-sharing blocks co-resident
  dendrite_gemm<<<grid1, 256, 0, stream>>>(x, dW, dB, dend);

  soma_kernel<<<(BATCH * N_NEURONS) / 256, 256, 0, stream>>>(dend, sW, sB, (float*)d_out);
}
